// SimHash_86088324481049
// MI455X (gfx1250) — matrix operand order repaired and checked by reading, not compiled, not run on hardware
//
#include <hip/hip_runtime.h>

typedef __bf16 v16bf __attribute__((ext_vector_type(16)));
typedef __bf16 v8bf  __attribute__((ext_vector_type(8)));
typedef __bf16 v4bf  __attribute__((ext_vector_type(4)));
typedef float  v8f   __attribute__((ext_vector_type(8)));
typedef float  v4f   __attribute__((ext_vector_type(4)));

#define D        256
#define BITS     24
#define TILE_M   128        // rows per block (8 waves x 16 rows)
#define KCHUNK   128        // K staged per pass (2 passes cover D=256)
#define XPITCH   136        // bf16 units; 272B row stride -> 68 dwords -> conflict-free
#define BPITCH   272        // bf16 units; 544B row stride, 32B-aligned row starts

__global__ __launch_bounds__(256)
void simhash_wmma_kernel(const float* __restrict__ x,
                         const float* __restrict__ rm,
                         const int*   __restrict__ bset,
                         float*       __restrict__ out)
{
    __shared__ __attribute__((aligned(32))) __bf16 Blds[32 * BPITCH];
    __shared__ __attribute__((aligned(32))) __bf16 Xlds[TILE_M * XPITCH];

    const int tid  = threadIdx.x;
    const int lane = tid & 31;
    const int wv   = tid >> 5;          // wave id 0..7
    const int lr   = lane & 15;         // lane within 16-group
    const int hl   = lane >> 4;         // half: 0 = lanes 0-15, 1 = lanes 16-31
    const int blockRow = blockIdx.x * TILE_M;

    // ---- Stage B: random_matrix [256 x 24] -> LDS transposed [n][k] bf16,
    //      n padded to 32 with zeros (padded cols give +0 products -> sign 0).
    for (int i = tid; i < 32 * D; i += 256) {
        int n = i >> 8;                 // 0..31
        int k = i & (D - 1);            // 0..255
        float v = (n < BITS) ? rm[k * BITS + n] : 0.0f;
        Blds[n * BPITCH + k] = (__bf16)v;
    }

    v8f c0 = {};                        // cols 0..15
    v8f c1 = {};                        // cols 16..31 (16..23 valid)

    for (int kc = 0; kc < D / KCHUNK; ++kc) {
        __syncthreads();
        // ---- Stage 128x128 fp32 chunk of x -> bf16 LDS (coalesced float4 loads)
        #pragma unroll
        for (int i = 0; i < 16; ++i) {
            int idx = tid + i * 256;                 // 0..4095
            int row = idx >> 5;                      // 0..127
            int c4  = idx & 31;                      // 0..31 (float4 index)
            const float* gp = x + (size_t)(blockRow + row) * D + kc * KCHUNK + c4 * 4;
            if (kc == 0)                             // warm next chunk: global_prefetch_b8
                __builtin_prefetch(gp + KCHUNK, 0, 0);
            v4f f = *(const v4f*)gp;
            v4bf h;
            h[0] = (__bf16)f[0]; h[1] = (__bf16)f[1];
            h[2] = (__bf16)f[2]; h[3] = (__bf16)f[3];
            *(v4bf*)(&Xlds[row * XPITCH + c4 * 4]) = h;
        }
        __syncthreads();

        const int arow = wv * 16 + lr;               // this lane's A row in tile
        #pragma unroll
        for (int ks = 0; ks < KCHUNK / 32; ++ks) {
            // A fragment: 16x32 bf16 (lanes0-15 K[0..15];
            //             lanes16-31 K[16..31])
            int k0 = ks * 32 + hl * 16;
            union { v16bf v; v8bf h[2]; } a;
            a.h[0] = *(const v8bf*)(&Xlds[arow * XPITCH + k0]);
            a.h[1] = *(const v8bf*)(&Xlds[arow * XPITCH + k0 + 8]);

            // B fragments: 32x16 bf16 (lane holds column n, 16 contiguous K;
            //              lanes0-15 K lo-half, lanes16-31 K hi-half)
            int kb = kc * KCHUNK + ks * 32 + hl * 16;
            union { v16bf v; v8bf h[2]; } b0, b1;
            b0.h[0] = *(const v8bf*)(&Blds[lr * BPITCH + kb]);
            b0.h[1] = *(const v8bf*)(&Blds[lr * BPITCH + kb + 8]);
            b1.h[0] = *(const v8bf*)(&Blds[(16 + lr) * BPITCH + kb]);
            b1.h[1] = *(const v8bf*)(&Blds[(16 + lr) * BPITCH + kb + 8]);

            c0 = __builtin_amdgcn_wmma_f32_16x16x32_bf16(
                     false, a.v, false, b0.v, (short)0, c0, false, false);
            c1 = __builtin_amdgcn_wmma_f32_16x16x32_bf16(
                     false, a.v, false, b1.v, (short)0, c1, false, false);
        }
    }

    // ---- Epilogue: sign bits -> 24-bit hash per row -> table gather -> bool
    // C layout: VGPR r, lanes0-15: M=r,N=lane; lanes16-31: M=8+r,N=lane-16
    #pragma unroll
    for (int r = 0; r < 8; ++r) {
        unsigned s0 = __float_as_uint(c0[r]) >> 31;
        unsigned s1 = __float_as_uint(c1[r]) >> 31;
        unsigned m  = (s0 << lr) | (s1 << (16 + lr));
        m |= __shfl_xor(m, 1);
        m |= __shfl_xor(m, 2);
        m |= __shfl_xor(m, 4);
        m |= __shfl_xor(m, 8);
        m &= 0x00FFFFFFu;
        if (lr == 0) {
            int row = blockRow + wv * 16 + hl * 8 + r;
            int bv  = bset[m >> 3];
            out[row] = ((bv >> (m & 7)) & 1) ? 1.0f : 0.0f;
        }
    }
}

extern "C" void kernel_launch(void* const* d_in, const int* in_sizes, int n_in,
                              void* d_out, int out_size, void* d_ws, size_t ws_size,
                              hipStream_t stream) {
    const float* x   = (const float*)d_in[0];   // [B, 256] fp32
    const float* rm  = (const float*)d_in[1];   // [256, 24] fp32
    const int*   bs  = (const int*)d_in[2];     // [2^21] int32 byte values
    float* out = (float*)d_out;                 // [B] bool-as-float

    const int Bn = in_sizes[0] / D;             // 131072
    dim3 grid(Bn / TILE_M), block(256);
    hipLaunchKernelGGL(simhash_wmma_kernel, grid, block, 0, stream, x, rm, bs, out);
}
